// RelPositionMultiHeadedAttention_67267777790071
// MI455X (gfx1250) — compile-verified, matrix operand order checked by reading, not run on hardware
//
#include <hip/hip_runtime.h>
#include <hip/hip_bf16.h>
#include <math.h>

#define BB 8
#define TT 1024
#define FF 512
#define HH 8
#define DK 64
#define LL 2047

typedef __attribute__((ext_vector_type(16))) __bf16 v16bf;
typedef __attribute__((ext_vector_type(8)))  float  v8f;
typedef __attribute__((ext_vector_type(16))) float  v16f;
typedef __attribute__((ext_vector_type(4)))  float  v4f;

#define NEG_INF (-__builtin_inff())

static __device__ __forceinline__ v8f wmma_bf16(v16bf a, v16bf b, v8f c) {
  // D = A(16x32 bf16) x B(32x16 bf16) + C(16x16 f32)
  return __builtin_amdgcn_wmma_f32_16x16x32_bf16(false, a, false, b, (short)0, c, false, false);
}

static __device__ __forceinline__ v16bf cvt16(v16f x) {
  v16bf r;
  #pragma unroll
  for (int j = 0; j < 16; ++j) r[j] = (__bf16)x[j];
  return r;
}

// ---------------------------------------------------------------------------
// Weight prep: Wt[n*512+k] = bf16(W[k*512+n])  (512x512)
// Makes every WMMA B-fragment a single contiguous 32B load.
// ---------------------------------------------------------------------------
__global__ __launch_bounds__(256)
void transpose_w_bf16(const float* __restrict__ W, __bf16* __restrict__ Wt)
{
  const int tid = blockIdx.x * 256 + threadIdx.x;   // 512*128 threads
  const int k4  = (tid & 127) * 4;
  const int n   = tid >> 7;
  #pragma unroll
  for (int i = 0; i < 4; ++i)
    Wt[(size_t)n * FF + k4 + i] = (__bf16)W[(size_t)(k4 + i) * FF + n];
}

// ---------------------------------------------------------------------------
// C[M,512] = A[M,512] @ W[512,512] (+bias); A fp32, W pre-transposed bf16.
// layout 0: fp32 row-major store (C).
// layout 1: KV epilogue: fp32 scatter into cache (B,H,S,2*DK) at koff, plus
//           bf16 staging copy: koff==0 -> K (B,H,S,DK); koff==DK -> V^T
//           (B,H,DK,S) for column-major B-fragment reads.
// layout 2: bf16 row-major store only (Cbf) -- pos projection.
// One wave computes a 16x64 tile; 4 waves per block.
// ---------------------------------------------------------------------------
__global__ __launch_bounds__(128)
void gemm512_bf16(const float* __restrict__ A, const __bf16* __restrict__ Wt,
                  const float* __restrict__ bias, float* __restrict__ C,
                  __bf16* __restrict__ Cbf, int M, int Mtiles, int layout, int koff)
{
  const int wid = blockIdx.x * 4 + (threadIdx.x >> 5);
  if (wid >= Mtiles * 8) return;                 // wave-uniform exit
  const int lane = threadIdx.x & 31;
  const int half = lane >> 4;                    // which 16 of the K=32 chunk
  const int lr   = lane & 15;                    // row (A) / col (B) in tile
  const int mt = wid >> 3;
  const int n0 = (wid & 7) * 64;
  const int r  = mt * 16 + lr;
  const int rc = r < M ? r : (M - 1);            // clamp loads for ragged M
  const float* arow = A + (size_t)rc * FF;
  const __bf16* wb0 = Wt + (size_t)(n0 + lr) * FF;

  v8f acc0 = {}, acc1 = {}, acc2 = {}, acc3 = {};
  for (int kc = 0; kc < FF; kc += 32) {
    const int kb = kc + half * 16;
    const v16bf a  = cvt16(*(const v16f*)(arow + kb));
    const v16bf b0 = *(const v16bf*)(wb0 + kb);
    const v16bf b1 = *(const v16bf*)(wb0 + 16 * FF + kb);
    const v16bf b2 = *(const v16bf*)(wb0 + 32 * FF + kb);
    const v16bf b3 = *(const v16bf*)(wb0 + 48 * FF + kb);
    acc0 = wmma_bf16(a, b0, acc0);
    acc1 = wmma_bf16(a, b1, acc1);
    acc2 = wmma_bf16(a, b2, acc2);
    acc3 = wmma_bf16(a, b3, acc3);
  }

  #pragma unroll
  for (int nt = 0; nt < 4; ++nt) {
    v8f acc = nt == 0 ? acc0 : nt == 1 ? acc1 : nt == 2 ? acc2 : acc3;
    #pragma unroll
    for (int rr = 0; rr < 8; ++rr) {
      const int m = rr + 8 * half;               // C layout: lanes16-31 hold M+8
      const int grow = mt * 16 + m;
      if (grow >= M) continue;
      const int gcol = n0 + nt * 16 + lr;
      const float v = acc[rr] + (bias ? bias[gcol] : 0.f);
      if (layout == 0) {
        C[(size_t)grow * FF + gcol] = v;
      } else if (layout == 1) {
        const int b = grow >> 10, t = grow & (TT - 1);
        const int hh = gcol >> 6, dd = gcol & 63;
        const size_t bh = (size_t)(b * HH + hh);
        C[(bh * TT + t) * (2 * DK) + koff + dd] = v;        // fp32 new_cache
        if (koff == 0) Cbf[(bh * TT + t) * DK + dd] = (__bf16)v;   // K bf16
        else           Cbf[(bh * DK + dd) * TT + t] = (__bf16)v;   // V^T bf16
      } else {
        Cbf[(size_t)grow * FF + gcol] = (__bf16)v;          // pos bf16
      }
    }
  }
}

// ---------------------------------------------------------------------------
// Fused rel-pos attention. One wave32 workgroup handles 16 query rows of one
// (b,h). Full score row (16x1024 f32 = 64KB) lives in LDS; rel_shift is done
// with lane permutes on the unshifted q_v @ p^T WMMA tiles.
// ---------------------------------------------------------------------------
__global__ __launch_bounds__(32)
void relattn_fused(const float* __restrict__ qbuf, const __bf16* __restrict__ pbf,
                   const __bf16* __restrict__ kbf, const __bf16* __restrict__ vbft,
                   const float* __restrict__ bias_u, const float* __restrict__ bias_v,
                   const int* __restrict__ mask, float* __restrict__ xbuf)
{
  __shared__ float sc[16][TT];                   // 64 KB of 320 KB/WGP

  const int bh = blockIdx.x >> 6;                // 64 row-tiles per (b,h)
  const int tb = blockIdx.x & 63;
  const int b = bh >> 3, h = bh & 7;
  const int lane = threadIdx.x;
  const int half = lane >> 4, lr = lane & 15;
  const int t0 = tb * 16;

  // q fragments with bias_u / bias_v and the 1/sqrt(Dk)=0.125 scale folded in.
  v16bf au0, au1, av0, av1;
  {
    const float* qrow = qbuf + ((size_t)(b * TT + t0 + lr)) * FF + h * DK;
    const v16f q0  = *(const v16f*)(qrow + half * 16);
    const v16f q1  = *(const v16f*)(qrow + 32 + half * 16);
    const v16f bu0 = *(const v16f*)(bias_u + h * DK + half * 16);
    const v16f bu1 = *(const v16f*)(bias_u + h * DK + 32 + half * 16);
    const v16f bv0 = *(const v16f*)(bias_v + h * DK + half * 16);
    const v16f bv1 = *(const v16f*)(bias_v + h * DK + 32 + half * 16);
    #pragma unroll
    for (int j = 0; j < 16; ++j) {
      au0[j] = (__bf16)((q0[j] + bu0[j]) * 0.125f);
      au1[j] = (__bf16)((q1[j] + bu1[j]) * 0.125f);
      av0[j] = (__bf16)((q0[j] + bv0[j]) * 0.125f);
      av1[j] = (__bf16)((q1[j] + bv1[j]) * 0.125f);
    }
  }
  const __bf16* kbase = kbf  + (size_t)(b * HH + h) * TT * DK;
  const __bf16* vbase = vbft + (size_t)(b * HH + h) * DK * TT;
  const int*    mrow  = mask + b * TT;

  // ---- Pass 1: scores into LDS -------------------------------------------
  for (int s0 = 0; s0 < TT; s0 += 32) {
    v8f ac0 = {}, ac1 = {};
    {
      const __bf16* kr0 = kbase + (size_t)(s0 + lr) * DK + half * 16;
      const __bf16* kr1 = kbase + (size_t)(s0 + 16 + lr) * DK + half * 16;
      const v16bf k00 = *(const v16bf*)(kr0);
      const v16bf k01 = *(const v16bf*)(kr0 + 32);
      const v16bf k10 = *(const v16bf*)(kr1);
      const v16bf k11 = *(const v16bf*)(kr1 + 32);
      ac0 = wmma_bf16(au0, k00, ac0); ac0 = wmma_bf16(au1, k01, ac0);
      ac1 = wmma_bf16(au0, k10, ac1); ac1 = wmma_bf16(au1, k11, ac1);
    }

    // bd window: p rows [d0-15, d0+32], where d0 = T-1 + s0 - t0.
    // bd_shifted[m,n] = bd_raw[m, 15+n-m] -> same VGPR slot, lane permute.
    const int d0 = (TT - 1) + s0 - t0;
    v8f bdt0 = {}, bdt1 = {}, bdt2 = {};
    #pragma unroll
    for (int jt = 0; jt < 3; ++jt) {
      int d = d0 - 15 + jt * 16 + lr;
      d = d < 0 ? 0 : (d > LL - 1 ? LL - 1 : d);   // clamped rows never read
      const __bf16* prow = pbf + (size_t)d * FF + h * DK + half * 16;
      const v16bf p0 = *(const v16bf*)(prow);
      const v16bf p1 = *(const v16bf*)(prow + 32);
      v8f t = {};
      t = wmma_bf16(av0, p0, t);
      t = wmma_bf16(av1, p1, t);
      if (jt == 0) bdt0 = t; else if (jt == 1) bdt1 = t; else bdt2 = t;
    }

    const int mv0 = mrow[s0 + lr];
    const int mv1 = mrow[s0 + 16 + lr];
    #pragma unroll
    for (int rr = 0; rr < 8; ++rr) {
      const int m = rr + 8 * half;
      const int jj0 = 15 + lr - m;                 // in [0,30]  -> tile 0/1
      const int jj1 = 31 + lr - m;                 // in [16,46] -> tile 1/2
      const int sl0 = (jj0 & 15) + 16 * half;      // bd row m lives in same half
      const int sl1 = (jj1 & 15) + 16 * half;
      const float c00 = __shfl(bdt0[rr], sl0, 32);
      const float c01 = __shfl(bdt1[rr], sl0, 32);
      const float c10 = __shfl(bdt1[rr], sl1, 32);
      const float c11 = __shfl(bdt2[rr], sl1, 32);
      const float bd0 = (jj0 < 16) ? c00 : c01;
      const float bd1 = (jj1 < 32) ? c10 : c11;
      const float v0 = ac0[rr] + bd0;
      const float v1 = ac1[rr] + bd1;
      sc[m][s0 + lr]      = mv0 ? v0 : NEG_INF;
      sc[m][s0 + 16 + lr] = mv1 ? v1 : NEG_INF;
    }
  }
  __syncthreads();

  // ---- Pass 2: softmax (lane (lr,half) owns row lr, half the cols) --------
  {
    v4f* row = (v4f*)&sc[lr][half * 512];
    float mx = NEG_INF;
    for (int i = 0; i < 128; ++i) {
      const v4f x = row[i];
      mx = fmaxf(mx, fmaxf(fmaxf(x[0], x[1]), fmaxf(x[2], x[3])));
    }
    mx = fmaxf(mx, __shfl_xor(mx, 16, 32));
    float sum = 0.f;
    for (int i = 0; i < 128; ++i) {
      v4f x = row[i];
      #pragma unroll
      for (int c = 0; c < 4; ++c) { x[c] = __expf(x[c] - mx); sum += x[c]; }
      row[i] = x;
    }
    sum += __shfl_xor(sum, 16, 32);
    const float inv = 1.f / sum;
    for (int i = 0; i < 128; ++i) row[i] *= inv;
  }
  __syncthreads();

  // ---- Pass 3: O = P @ V --------------------------------------------------
  v8f o0 = {}, o1 = {}, o2 = {}, o3 = {};
  for (int s0 = 0; s0 < TT; s0 += 32) {
    const v16bf ap = cvt16(*(const v16f*)(&sc[lr][s0 + half * 16]));
    const __bf16* vp = vbase + (size_t)lr * TT + s0 + half * 16;
    const v16bf b0 = *(const v16bf*)(vp);
    const v16bf b1 = *(const v16bf*)(vp + 16 * TT);
    const v16bf b2 = *(const v16bf*)(vp + 32 * TT);
    const v16bf b3 = *(const v16bf*)(vp + 48 * TT);
    o0 = wmma_bf16(ap, b0, o0);
    o1 = wmma_bf16(ap, b1, o1);
    o2 = wmma_bf16(ap, b2, o2);
    o3 = wmma_bf16(ap, b3, o3);
  }

  float* xrow = xbuf + (size_t)(b * TT + t0) * FF + h * DK;
  #pragma unroll
  for (int rr = 0; rr < 8; ++rr) {
    const int m = rr + 8 * half;
    float* xr = xrow + (size_t)m * FF + lr;
    xr[0]  = o0[rr];
    xr[16] = o1[rr];
    xr[32] = o2[rr];
    xr[48] = o3[rr];
  }
}

// ---------------------------------------------------------------------------
extern "C" void kernel_launch(void* const* d_in, const int* in_sizes, int n_in,
                              void* d_out, int out_size, void* d_ws, size_t ws_size,
                              hipStream_t stream) {
  const float* query  = (const float*)d_in[0];
  const float* key_in = (const float*)d_in[1];
  const float* value  = (const float*)d_in[2];
  const float* pos    = (const float*)d_in[3];
  const float* Wq = (const float*)d_in[4];   const float* bq = (const float*)d_in[5];
  const float* Wk = (const float*)d_in[6];   const float* bk = (const float*)d_in[7];
  const float* Wv = (const float*)d_in[8];   const float* bv = (const float*)d_in[9];
  const float* Wp = (const float*)d_in[10];
  const float* Wo = (const float*)d_in[11];  const float* bo = (const float*)d_in[12];
  const float* bu  = (const float*)d_in[13];
  const float* bvv = (const float*)d_in[14];
  const int*   mask = (const int*)d_in[15];

  float* out   = (float*)d_out;                       // (B,T,F)
  float* cache = out + (size_t)BB * TT * FF;          // (B,H,S,2*DK) fp32

  // workspace layout
  float*  qbuf = (float*)d_ws;                                  // 16 MB fp32
  float*  xbuf = qbuf + (size_t)BB * TT * FF;                   // 16 MB fp32
  __bf16* kbf  = (__bf16*)(xbuf + (size_t)BB * TT * FF);        //  8 MB (B,H,S,DK)
  __bf16* vbft = kbf  + (size_t)BB * HH * TT * DK;              //  8 MB (B,H,DK,S)
  __bf16* pbf  = vbft + (size_t)BB * HH * TT * DK;              //  2 MB (L,F) padded
  __bf16* wt   = pbf  + (size_t)2048 * FF;                      //  2.5 MB: 5 weights
  __bf16* wq_t = wt;
  __bf16* wk_t = wt + 1 * (size_t)FF * FF;
  __bf16* wv_t = wt + 2 * (size_t)FF * FF;
  __bf16* wp_t = wt + 3 * (size_t)FF * FF;
  __bf16* wo_t = wt + 4 * (size_t)FF * FF;

  const int MQ = BB * TT;                             // 8192
  dim3 blk(128);

  // weight prep (tiny)
  transpose_w_bf16<<<256, 256, 0, stream>>>(Wq, wq_t);
  transpose_w_bf16<<<256, 256, 0, stream>>>(Wk, wk_t);
  transpose_w_bf16<<<256, 256, 0, stream>>>(Wv, wv_t);
  transpose_w_bf16<<<256, 256, 0, stream>>>(Wp, wp_t);
  transpose_w_bf16<<<256, 256, 0, stream>>>(Wo, wo_t);

  gemm512_bf16<<<512 * 8 / 4, blk, 0, stream>>>(query,  wq_t, bq,      qbuf, nullptr, MQ, 512, 0, 0);
  gemm512_bf16<<<512 * 8 / 4, blk, 0, stream>>>(key_in, wk_t, bk,      cache, kbf,    MQ, 512, 1, 0);
  gemm512_bf16<<<512 * 8 / 4, blk, 0, stream>>>(value,  wv_t, bv,      cache, vbft,   MQ, 512, 1, DK);
  gemm512_bf16<<<128 * 8 / 4, blk, 0, stream>>>(pos,    wp_t, nullptr, nullptr, pbf,  LL, 128, 2, 0);

  relattn_fused<<<BB * HH * (TT / 16), dim3(32), 0, stream>>>(qbuf, pbf, kbf, vbft,
                                                              bu, bvv, mask, xbuf);

  gemm512_bf16<<<512 * 8 / 4, blk, 0, stream>>>(xbuf, wo_t, bo, out, nullptr, MQ, 512, 0, 0);
}
